// MultiOutputNet_63780264345754
// MI455X (gfx1250) — matrix operand order repaired and checked by reading, not compiled, not run on hardware
//
#include <hip/hip_runtime.h>
#include <hip/hip_bf16.h>

// ---------------------------------------------------------------------------
// MultiOutputNet on MI455X (gfx1250): fused bf16 WMMA pipeline.
//   shared = relu(x @ Ws + bs)                       [2048,512]
//   per branch n (256): h1=relu(shared@W1[n]) -> h2=relu(h1@W2[n]) -> h2@W3[n]
// Compute-bound (~780 FLOP/byte vs ~100 machine balance) => everything on the
// v_wmma_f32_16x16x32_bf16 pipe; h1/h2 never touch HBM (fused in LDS).
// This rev: explicit 2-deep software pipeline on B fragments (double-buffered
// registers) so waits become loadcnt<=1 instead of loadcnt==0.
// ---------------------------------------------------------------------------

#define Bsz    2048
#define IN_DIM 700
#define IN_PAD 704   // pad K of first GEMM to multiple of 32
#define SHD    512
#define L1D    128
#define L2D    64
#define OUTD   9
#define OUTP   16    // pad last N to one 16-wide WMMA tile
#define NBR    256

typedef __attribute__((ext_vector_type(16))) __bf16 v16bf;
typedef __attribute__((ext_vector_type(8)))  float  v8f;

union Frag16 {          // one 16x32 A or 32x16 B bf16 fragment: 16 bf16 / lane
  v16bf v;
  uint4 q[2];
};

__device__ __forceinline__ unsigned int f32_to_bf16_bits(float f) {
  unsigned int u = __float_as_uint(f);
  return (u + 0x7FFFu + ((u >> 16) & 1u)) >> 16;   // round-to-nearest-even
}

__device__ __forceinline__ v8f wmma_bf16(const Frag16& a, const Frag16& b, v8f c) {
  // (neg_a, A, neg_b, B, c_mod, C, reuse_a, reuse_b)
  return __builtin_amdgcn_wmma_f32_16x16x32_bf16(false, a.v, false, b.v,
                                                 (short)0, c, false, false);
}

// ---------------------------------------------------------------------------
// Pack x [2048,700] f32 -> row-major bf16 [2048,704], zero K-padding.
// ---------------------------------------------------------------------------
__global__ void pack_x_kernel(const float* __restrict__ x,
                              unsigned short* __restrict__ xb) {
  int idx = blockIdx.x * blockDim.x + threadIdx.x;
  if (idx >= Bsz * IN_PAD) return;
  int row = idx / IN_PAD, c = idx - row * IN_PAD;
  unsigned short v = 0;
  if (c < IN_DIM) v = (unsigned short)f32_to_bf16_bits(x[(size_t)row * IN_DIM + c]);
  xb[idx] = v;
}

// ---------------------------------------------------------------------------
// Pack a (possibly branch-stacked) row-major [K,N] f32 weight into the exact
// wave32 B-fragment layout of V_WMMA_*_16X16X32_BF16:
//   fragment (nt,kt): 256 dwords; lane = khalf*16 + (n%16); dword j holds
//   bf16 pair (k, k+1) with k = kt*32 + (j/4)*16 + khalf*8 + 2*(j%4).
// ---------------------------------------------------------------------------
__global__ void pack_b_kernel(const float* __restrict__ src,
                              unsigned int* __restrict__ dst,
                              int K, int N, int KT,
                              long long dwPerBranch, long long totalDw) {
  long long d = (long long)blockIdx.x * blockDim.x + threadIdx.x;
  if (d >= totalDw) return;
  long long br  = d / dwPerBranch;
  int rem  = (int)(d - br * dwPerBranch);
  int j    = rem & 7;
  int ln   = (rem >> 3) & 31;
  int frag = rem >> 8;
  int kt   = frag % KT;
  int nt   = frag / KT;
  int k0   = kt * 32 + ((j >> 2) << 4) + ((ln >> 4) << 3) + ((j & 3) << 1);
  int n    = (nt << 4) + (ln & 15);
  const float* s = src + (size_t)br * K * N;
  unsigned int lo = 0, hi = 0;
  if (n < N) {
    if (k0     < K) lo = f32_to_bf16_bits(s[(size_t)k0       * N + n]);
    if (k0 + 1 < K) hi = f32_to_bf16_bits(s[(size_t)(k0 + 1) * N + n]);
  }
  dst[d] = lo | (hi << 16);
}

// A-fragment from row-major bf16, base = uint4 ptr at (row, khalf*8):
// chunk0 at uint4 idx kt*4 (+khalf already in base), chunk1 at +2.
__device__ __forceinline__ void load_a(Frag16& a, const uint4* abase, int kt) {
  a.q[0] = abase[kt * 4];
  a.q[1] = abase[kt * 4 + 2];
}
// B-fragment from packed stream, base = uint4 ptr + lane*2; frag stride 64 uint4.
__device__ __forceinline__ void load_b(Frag16& b, const uint4* bbase, int frag) {
  b.q[0] = bbase[frag * 64];
  b.q[1] = bbase[frag * 64 + 1];
}

// ---------------------------------------------------------------------------
// shared = relu(x @ Ws + bs) -> bf16 [2048,512].
// Block = 128 threads (4 waves); tile 128(M) x 128(N); each wave: 32 rows x 8
// N-tiles, K = 704 (22 WMMA steps, fully unrolled, 2-deep B pipeline).
// ---------------------------------------------------------------------------
__global__ void __launch_bounds__(128)
shared_gemm_kernel(const unsigned short* __restrict__ xb,
                   const unsigned int* __restrict__ Wsb,
                   const float* __restrict__ bs,
                   unsigned short* __restrict__ shb) {
  const int lane  = threadIdx.x & 31;
  const int wave  = threadIdx.x >> 5;
  const int khalf = lane >> 4;
  const int rloc  = lane & 15;
  const int mloc  = khalf * 8;
  const int m0    = blockIdx.x * 128 + wave * 32;
  const int n0    = blockIdx.y * 128;
  const int KT    = IN_PAD / 32;       // 22

  v8f acc[2][8] = {};
  const uint4* a0 = (const uint4*)xb + (size_t)(m0 + rloc) * (IN_PAD / 8) + khalf;
  const uint4* a1 = a0 + 16 * (IN_PAD / 8);
  const uint4* bb = (const uint4*)Wsb + (size_t)(n0 >> 4) * KT * 64 + lane * 2;

  Frag16 A0, A1, A0n, A1n, Bcur, Bnxt;
  load_a(A0, a0, 0);
  load_a(A1, a1, 0);
  load_b(Bcur, bb, 0);
#pragma unroll
  for (int kt = 0; kt < KT; ++kt) {
    if (kt + 1 < KT) {
      load_a(A0n, a0, kt + 1);
      load_a(A1n, a1, kt + 1);
    }
#pragma unroll
    for (int nt = 0; nt < 8; ++nt) {
      const int nn = (nt + 1) & 7;
      const int nk = (nt == 7) ? kt + 1 : kt;
      if (nk < KT) load_b(Bnxt, bb, nn * KT + nk);
      acc[0][nt] = wmma_bf16(A0, Bcur, acc[0][nt]);
      acc[1][nt] = wmma_bf16(A1, Bcur, acc[1][nt]);
      Bcur = Bnxt;                    // renamed away under full unroll
    }
    A0 = A0n;
    A1 = A1n;
  }
#pragma unroll
  for (int nt = 0; nt < 8; ++nt) {
    const int col  = n0 + nt * 16 + rloc;
    const float bv = bs[col];
#pragma unroll
    for (int mt = 0; mt < 2; ++mt)
#pragma unroll
      for (int r = 0; r < 8; ++r) {
        float v = acc[mt][nt][r] + bv;
        v = v > 0.f ? v : 0.f;
        shb[(size_t)(m0 + mt * 16 + mloc + r) * SHD + col] =
            (unsigned short)f32_to_bf16_bits(v);
      }
  }
}

// ---------------------------------------------------------------------------
// Fully fused 3-layer branch pipeline. Grid = (256 branches, 16 batch tiles),
// block = 128 threads (4 waves x 32 batch rows). h1/h2 live in LDS only;
// each wave reads back exactly the strip it wrote (no cross-wave traffic).
// ---------------------------------------------------------------------------
__global__ void __launch_bounds__(128)
branch_kernel(const unsigned short* __restrict__ shb,
              const unsigned int* __restrict__ W1b,
              const float* __restrict__ b1,
              const unsigned int* __restrict__ W2b,
              const float* __restrict__ b2,
              const unsigned int* __restrict__ W3b,
              const float* __restrict__ b3,
              float* __restrict__ out) {
  __shared__ unsigned short h1[4][32][L1D];   // 32 KB
  __shared__ unsigned short h2[4][32][L2D];   // 16 KB

  const int lane   = threadIdx.x & 31;
  const int wave   = threadIdx.x >> 5;
  const int khalf  = lane >> 4;
  const int rloc   = lane & 15;
  const int mloc   = khalf * 8;
  const int branch = blockIdx.x;
  const int m0     = blockIdx.y * 128 + wave * 32;

  // ---- layer 1: h1 = relu(shared[32x512] @ W1[n][512x128]) ----
  {
    const int KT1 = SHD / 32;   // 16
    v8f acc[2][8] = {};
    const uint4* bb = (const uint4*)W1b + (size_t)branch * (8 * KT1 * 64) + lane * 2;
    const uint4* a0 = (const uint4*)shb + (size_t)(m0 + rloc) * (SHD / 8) + khalf;
    const uint4* a1 = a0 + 16 * (SHD / 8);

    Frag16 A0, A1, A0n, A1n, Bcur, Bnxt;
    load_a(A0, a0, 0);
    load_a(A1, a1, 0);
    load_b(Bcur, bb, 0);
#pragma unroll
    for (int kt = 0; kt < KT1; ++kt) {
      if (kt + 1 < KT1) {
        load_a(A0n, a0, kt + 1);
        load_a(A1n, a1, kt + 1);
      }
#pragma unroll
      for (int nt = 0; nt < 8; ++nt) {
        const int nn = (nt + 1) & 7;
        const int nk = (nt == 7) ? kt + 1 : kt;
        if (nk < KT1) load_b(Bnxt, bb, nn * KT1 + nk);
        acc[0][nt] = wmma_bf16(A0, Bcur, acc[0][nt]);
        acc[1][nt] = wmma_bf16(A1, Bcur, acc[1][nt]);
        Bcur = Bnxt;
      }
      A0 = A0n;
      A1 = A1n;
    }
#pragma unroll
    for (int nt = 0; nt < 8; ++nt) {
      const float bv = b1[branch * L1D + nt * 16 + rloc];
#pragma unroll
      for (int mt = 0; mt < 2; ++mt)
#pragma unroll
        for (int r = 0; r < 8; ++r) {
          float v = acc[mt][nt][r] + bv;
          v = v > 0.f ? v : 0.f;
          h1[wave][mt * 16 + mloc + r][nt * 16 + rloc] =
              (unsigned short)f32_to_bf16_bits(v);
        }
    }
  }
  __syncthreads();

  // ---- layer 2: h2 = relu(h1[32x128] @ W2[n][128x64]) ----
  {
    v8f acc[2][4] = {};
    const uint4* bb = (const uint4*)W2b + (size_t)branch * (4 * 4 * 64) + lane * 2;
    const uint4* a0 = (const uint4*)&h1[wave][rloc][0] + khalf;        // pitch 16 uint4
    const uint4* a1 = (const uint4*)&h1[wave][16 + rloc][0] + khalf;
#pragma unroll
    for (int kt = 0; kt < L1D / 32; ++kt) {          // 4
      Frag16 A0, A1;
      load_a(A0, a0, kt);
      load_a(A1, a1, kt);
#pragma unroll
      for (int nt = 0; nt < 4; ++nt) {
        Frag16 B;
        load_b(B, bb, nt * (L1D / 32) + kt);
        acc[0][nt] = wmma_bf16(A0, B, acc[0][nt]);
        acc[1][nt] = wmma_bf16(A1, B, acc[1][nt]);
      }
    }
#pragma unroll
    for (int nt = 0; nt < 4; ++nt) {
      const float bv = b2[branch * L2D + nt * 16 + rloc];
#pragma unroll
      for (int mt = 0; mt < 2; ++mt)
#pragma unroll
        for (int r = 0; r < 8; ++r) {
          float v = acc[mt][nt][r] + bv;
          v = v > 0.f ? v : 0.f;
          h2[wave][mt * 16 + mloc + r][nt * 16 + rloc] =
              (unsigned short)f32_to_bf16_bits(v);
        }
    }
  }
  __syncthreads();

  // ---- layer 3: out = h2[32x64] @ W3[n][64x16(pad)] + b3, keep 9 cols ----
  {
    v8f acc[2] = {};
    const uint4* bb = (const uint4*)W3b + (size_t)branch * (1 * 2 * 64) + lane * 2;
    const uint4* a0 = (const uint4*)&h2[wave][rloc][0] + khalf;        // pitch 8 uint4
    const uint4* a1 = (const uint4*)&h2[wave][16 + rloc][0] + khalf;
#pragma unroll
    for (int kt = 0; kt < L2D / 32; ++kt) {          // 2
      Frag16 A0, A1;
      A0.q[0] = a0[kt * 4]; A0.q[1] = a0[kt * 4 + 2];
      A1.q[0] = a1[kt * 4]; A1.q[1] = a1[kt * 4 + 2];
      Frag16 B;
      load_b(B, bb, kt);
      acc[0] = wmma_bf16(A0, B, acc[0]);
      acc[1] = wmma_bf16(A1, B, acc[1]);
    }
    if (rloc < OUTD) {
      const float bv = b3[branch * OUTD + rloc];
#pragma unroll
      for (int mt = 0; mt < 2; ++mt)
#pragma unroll
        for (int r = 0; r < 8; ++r) {
          const int brow = m0 + mt * 16 + mloc + r;
          out[((size_t)brow * NBR + branch) * OUTD + rloc] = acc[mt][r] + bv;
        }
    }
  }
}

// ---------------------------------------------------------------------------
extern "C" void kernel_launch(void* const* d_in, const int* in_sizes, int n_in,
                              void* d_out, int out_size, void* d_ws, size_t ws_size,
                              hipStream_t stream) {
  const float* x  = (const float*)d_in[0];
  const float* Ws = (const float*)d_in[1];
  const float* bs = (const float*)d_in[2];
  const float* W1 = (const float*)d_in[3];
  const float* b1 = (const float*)d_in[4];
  const float* W2 = (const float*)d_in[5];
  const float* b2 = (const float*)d_in[6];
  const float* W3 = (const float*)d_in[7];
  const float* b3 = (const float*)d_in[8];
  float* out = (float*)d_out;

  char* ws = (char*)d_ws;
  size_t off = 0;
  unsigned short* xb  = (unsigned short*)(ws + off); off += (size_t)Bsz * IN_PAD * 2;
  unsigned int*   Wsb = (unsigned int*)(ws + off);   off += (size_t)(SHD/16)*(IN_PAD/32)*256*4;
  unsigned short* shb = (unsigned short*)(ws + off); off += (size_t)Bsz * SHD * 2;
  unsigned int*   W1b = (unsigned int*)(ws + off);   off += (size_t)NBR*(L1D/16)*(SHD/32)*256*4;
  unsigned int*   W2b = (unsigned int*)(ws + off);   off += (size_t)NBR*(L2D/16)*(L1D/32)*256*4;
  unsigned int*   W3b = (unsigned int*)(ws + off);   off += (size_t)NBR*(OUTP/16)*(L2D/32)*256*4;
  (void)off; (void)ws_size; (void)in_sizes; (void)n_in; (void)out_size;

  // -- pack activations + weights into WMMA fragment layouts --
  {
    int total = Bsz * IN_PAD;
    pack_x_kernel<<<(total + 255) / 256, 256, 0, stream>>>(x, xb);
  }
  {
    long long dpb = (long long)(SHD/16) * (IN_PAD/32) * 256;
    pack_b_kernel<<<(unsigned)((dpb + 255) / 256), 256, 0, stream>>>(
        Ws, Wsb, IN_DIM, SHD, IN_PAD/32, dpb, dpb);
  }
  {
    long long dpb = (long long)(L1D/16) * (SHD/32) * 256, tot = dpb * NBR;
    pack_b_kernel<<<(unsigned)((tot + 255) / 256), 256, 0, stream>>>(
        W1, W1b, SHD, L1D, SHD/32, dpb, tot);
  }
  {
    long long dpb = (long long)(L2D/16) * (L1D/32) * 256, tot = dpb * NBR;
    pack_b_kernel<<<(unsigned)((tot + 255) / 256), 256, 0, stream>>>(
        W2, W2b, L1D, L2D, L1D/32, dpb, tot);
  }
  {
    long long dpb = (long long)(OUTP/16) * (L2D/32) * 256, tot = dpb * NBR;
    pack_b_kernel<<<(unsigned)((tot + 255) / 256), 256, 0, stream>>>(
        W3, W3b, L2D, OUTD, L2D/32, dpb, tot);
  }

  // -- shared layer GEMM --
  shared_gemm_kernel<<<dim3(Bsz / 128, SHD / 128), 128, 0, stream>>>(xb, Wsb, bs, shb);

  // -- fused 256-branch 3-layer pipeline --
  branch_kernel<<<dim3(NBR, Bsz / 128), 128, 0, stream>>>(
      shb, W1b, b1, W2b, b2, W3b, b3, out);
}
